// Net_38577396253444
// MI455X (gfx1250) — matrix operand order repaired and checked by reading, not compiled, not run on hardware
//
#include <hip/hip_runtime.h>
#include <hip/hip_bf16.h>
#include <math.h>

typedef __attribute__((ext_vector_type(16))) _Float16 v16h;
typedef __attribute__((ext_vector_type(8)))  _Float16 v8h;
typedef __attribute__((ext_vector_type(8)))  float    v8f;
typedef __attribute__((ext_vector_type(4)))  float    f4v;
// float4 view with only 4-byte alignment guarantee (legal for global b128 in DWORD mode)
typedef f4v uf4 __attribute__((aligned(4)));

// ---------------------------------------------------------------------------
// Weight prep: pad Cout -> Mpad rows and convert fp32 -> fp16 once per layer.
// Removes the m<Cout guard and the A-side cvt from the conv hot loop.
// ---------------------------------------------------------------------------
__global__ __launch_bounds__(256) void prep_weights_kernel(
    const float* __restrict__ w, _Float16* __restrict__ wh,
    int Cout, int Mpad, int K)
{
    const int i = blockIdx.x * 256 + threadIdx.x;
    if (i >= Mpad * K) return;
    const int m = i / K;
    wh[i] = (m < Cout) ? (_Float16)w[i] : (_Float16)0.0f;
}

// ---------------------------------------------------------------------------
// Conv3D (4x4x4 VALID, stride 1) + bias + ReLU as implicit GEMM on WMMA.
// One wave computes a 16(out-chan) x 16(spatial) tile; K = Cin*64 stepped by 32
// through v_wmma_f32_16x16x32_f16 (f32 accumulate).
//
// Per k-step (branch-free):
//   A: two aligned b128 loads of 8 halfs (lane layout is K-contiguous per half)
//   B: four float4 row loads (fixed ci,kd per step; kh=e>>2, kw=e&3) + cvt
// ---------------------------------------------------------------------------
__global__ __launch_bounds__(32) void conv3d_wmma_kernel(
    const float* __restrict__ in, const _Float16* __restrict__ wh,
    const float* __restrict__ bias, float* __restrict__ out,
    int Cin, int Cout, int Din, int Hin, int Win,
    int Dout, int Hout, int Wout)
{
    const int lane = threadIdx.x;         // 0..31
    const int m16  = lane & 15;
    const int half = lane >> 4;
    const int HWo  = Hout * Wout;
    const int Nspat = Dout * HWo;
    const int K     = Cin * 64;

    const int co0 = blockIdx.y * 16;      // M tile base (within padded rows)
    const int n0  = blockIdx.x * 16;      // N (spatial) tile base

    // This lane's B column n = n0 + m16; clamp invalid columns to column 0's
    // patch (their D columns are never stored, so garbage values are fine).
    const int  n      = n0 + m16;
    const bool nvalid = (n < Nspat);
    const int  nc     = nvalid ? n : 0;
    const int  z0     = nc / HWo;
    const int  rem    = nc - z0 * HWo;
    const int  y0     = rem / Wout;
    const int  x0     = rem - y0 * Wout;

    const _Float16* wrow = wh + (size_t)(co0 + m16) * K;   // A row for this lane

    v8f acc = {};

    for (int k0 = 0; k0 < K; k0 += 32) {
        // ---- A fragment: a[0..7] <- K=k0+half*16+e ; a[8..15] <- K=k0+half*16+8+e
        const v8h a0 = *(const v8h*)(wrow + k0 + half * 16);
        const v8h a1 = *(const v8h*)(wrow + k0 + half * 16 + 8);
        v16h a;
        #pragma unroll
        for (int e = 0; e < 8; ++e) { a[e] = a0[e]; a[e + 8] = a1[e]; }

        // ---- B fragment: 16 consecutive K inside one 64-block -> fixed ci,kd
        const int kbase = k0 + half * 16;
        const int ci = kbase >> 6;
        const int kd = (kbase >> 4) & 3;
        const float* bp = in + ((size_t)(ci * Din + z0 + kd) * Hin + y0) * Win + x0;
        v16h b;
        #pragma unroll
        for (int kh = 0; kh < 4; ++kh) {
            const f4v row = *(const uf4*)(bp + kh * Win);
            b[kh * 4 + 0] = (_Float16)row.x;
            b[kh * 4 + 1] = (_Float16)row.y;
            b[kh * 4 + 2] = (_Float16)row.z;
            b[kh * 4 + 3] = (_Float16)row.w;
        }

        acc = __builtin_amdgcn_wmma_f32_16x16x32_f16(
            /*neg_a=*/false, a, /*neg_b=*/false, b,
            /*c_mod=*/(short)0, acc, /*reuse_a=*/false, /*reuse_b=*/false);
    }

    // ---- Store D: lane column n = n0+m16; VGPR r -> row m = co0 + half*8 + r.
    if (nvalid) {
        #pragma unroll
        for (int r = 0; r < 8; ++r) {
            const int m = co0 + half * 8 + r;
            if (m < Cout) {
                float v = acc[r] + bias[m];
                v = v > 0.0f ? v : 0.0f;                  // ReLU
                out[((size_t)(m * Dout + z0) * Hout + y0) * Wout + x0] = v;
            }
        }
    }
}

// ---------------------------------------------------------------------------
// 4x4x4 stride-1 VALID max-pool.
// ---------------------------------------------------------------------------
__global__ void maxpool3d_kernel(const float* __restrict__ in, float* __restrict__ out,
                                 int C, int Din, int Hin, int Win,
                                 int Dout, int Hout, int Wout)
{
    const int idx = blockIdx.x * blockDim.x + threadIdx.x;
    const int total = C * Dout * Hout * Wout;
    if (idx >= total) return;
    int t = idx;
    const int x = t % Wout; t /= Wout;
    const int y = t % Hout; t /= Hout;
    const int z = t % Dout; const int c = t / Dout;
    float m = -__builtin_inff();
    #pragma unroll
    for (int dz = 0; dz < 4; ++dz)
        #pragma unroll
        for (int dy = 0; dy < 4; ++dy) {
            const float* p = in + ((size_t)(c * Din + z + dz) * Hin + (y + dy)) * Win + x;
            #pragma unroll
            for (int dx = 0; dx < 4; ++dx) m = fmaxf(m, p[dx]);
        }
    out[idx] = m;
}

// ---------------------------------------------------------------------------
// Streaming matvec: out[row] = dot(W[row,:], h) + b[row], optional ReLU.
// One 256-thread block per row. This is the HBM-roofline stage
// (fc1_w = 411 MB fp32 ~= 17.6 us at 23.3 TB/s) -> keep it a pure stream.
// ---------------------------------------------------------------------------
__global__ __launch_bounds__(256) void fc_kernel(
    const float* __restrict__ W, const float* __restrict__ bvec,
    const float* __restrict__ h, float* __restrict__ out,
    int In, int relu)
{
    const int row = blockIdx.x;
    const float* Wr = W + (size_t)row * In;
    float s = 0.0f;
    for (int i = threadIdx.x; i < In; i += 256) {
        __builtin_prefetch(&Wr[i + 4096], 0, 1);   // global_prefetch_b8
        s += Wr[i] * h[i];
    }
    __shared__ float red[256];
    red[threadIdx.x] = s;
    __syncthreads();
    #pragma unroll
    for (int off = 128; off > 0; off >>= 1) {
        if (threadIdx.x < off) red[threadIdx.x] += red[threadIdx.x + off];
        __syncthreads();
    }
    if (threadIdx.x == 0) {
        float v = red[0] + bvec[row];
        if (relu) v = fmaxf(v, 0.0f);
        out[row] = v;
    }
}

// ---------------------------------------------------------------------------
// Head: sigmoid*25, box adjust (+/-2), 50x50 IoU NMS, masked outputs.
// Output: 200 floats boxes (adj*keep) then 50 floats scores*keep.
// ---------------------------------------------------------------------------
__global__ __launch_bounds__(64) void head_kernel(const float* __restrict__ v250,
                                                  float* __restrict__ out)
{
    __shared__ float adj[50][4];
    __shared__ float sc[50];
    const int t = threadIdx.x;
    if (t < 50) {
        float b[5];
        #pragma unroll
        for (int j = 0; j < 5; ++j)
            b[j] = 25.0f / (1.0f + expf(-v250[t * 5 + j]));
        adj[t][0] = b[0] - 2.0f;
        adj[t][1] = b[1] - 2.0f;
        adj[t][2] = b[2] + 2.0f;
        adj[t][3] = b[3] + 2.0f;
        sc[t]     = b[4] / 30.0f;
    }
    __syncthreads();
    if (t < 50) {
        const float ax0 = adj[t][0], ay0 = adj[t][1], ax1 = adj[t][2], ay1 = adj[t][3];
        const float areaA = (ax1 - ax0) * (ay1 - ay0);
        bool dominated = false;
        for (int j = 0; j < 50; ++j) {
            if (j == t) continue;
            const float bx0 = adj[j][0], by0 = adj[j][1], bx1 = adj[j][2], by1 = adj[j][3];
            const float ix0 = fmaxf(ax0, bx0), iy0 = fmaxf(ay0, by0);
            const float ix1 = fminf(ax1, bx1), iy1 = fminf(ay1, by1);
            const float inter = fmaxf(ix1 - ix0, 0.0f) * fmaxf(iy1 - iy0, 0.0f);
            const float areaB = (bx1 - bx0) * (by1 - by0);
            const float iou   = inter / (areaA + areaB - inter);
            if (iou > 0.05f && sc[j] >= sc[t]) dominated = true;
        }
        const bool keep = !dominated && !((ax0 > ax1) || (ay0 > ay1));
        const float kf = keep ? 1.0f : 0.0f;
        out[t * 4 + 0] = ax0 * kf;
        out[t * 4 + 1] = ay0 * kf;
        out[t * 4 + 2] = ax1 * kf;
        out[t * 4 + 3] = ay1 * kf;
        out[200 + t]   = sc[t] * kf;
    }
}

// ---------------------------------------------------------------------------
extern "C" void kernel_launch(void* const* d_in, const int* in_sizes, int n_in,
                              void* d_out, int out_size, void* d_ws, size_t ws_size,
                              hipStream_t stream) {
    (void)in_sizes; (void)n_in; (void)out_size; (void)ws_size;

    const float* x = (const float*)d_in[0];
    const float* cw[6]; const float* cb[6];
    for (int i = 0; i < 6; ++i) {
        cw[i] = (const float*)d_in[1 + 2 * i];
        cb[i] = (const float*)d_in[2 + 2 * i];
    }
    const float* fc1w = (const float*)d_in[13]; const float* fc1b = (const float*)d_in[14];
    const float* fc2w = (const float*)d_in[15]; const float* fc2b = (const float*)d_in[16];
    const float* fc7w = (const float*)d_in[17]; const float* fc7b = (const float*)d_in[18];

    // Scratch: two 4MB activation ping-pong buffers + 128KB padded f16 weights.
    float*     bufA = (float*)d_ws;
    float*     bufB = bufA + (1u << 20);
    _Float16*  wpad = (_Float16*)(bufA + 2 * (1u << 20));

    static const int CinL[6]  = {1, 4, 8, 16, 32, 32};
    static const int CoutL[6] = {4, 8, 16, 32, 32, 32};
    int Din = 37, Hin = 64, Win = 64;
    const float* cur = x;

    for (int L = 0; L < 6; ++L) {
        const int K    = CinL[L] * 64;
        const int Mpad = (CoutL[L] + 15) & ~15;
        const int Dc = Din - 3, Hc = Hin - 3, Wc = Win - 3;   // conv out dims
        const int Dp = Dc - 3,  Hp = Hc - 3,  Wp = Wc - 3;    // pool out dims
        const int Nspat = Dc * Hc * Wc;

        prep_weights_kernel<<<(Mpad * K + 255) / 256, 256, 0, stream>>>(
            cw[L], wpad, CoutL[L], Mpad, K);

        dim3 grid((Nspat + 15) / 16, Mpad / 16);
        conv3d_wmma_kernel<<<grid, 32, 0, stream>>>(
            cur, wpad, cb[L], bufA,
            CinL[L], CoutL[L], Din, Hin, Win, Dc, Hc, Wc);

        const int totalP = CoutL[L] * Dp * Hp * Wp;
        maxpool3d_kernel<<<(totalP + 255) / 256, 256, 0, stream>>>(
            bufA, bufB, CoutL[L], Dc, Hc, Wc, Dp, Hp, Wp);

        cur = bufB;
        Din = Dp; Hin = Hp; Win = Wp;
    }

    // bufB holds h (25088 floats, NCDHW flatten order == memory order).
    fc_kernel<<<4096, 256, 0, stream>>>(fc1w, fc1b, bufB, bufA, 25088, 1);
    fc_kernel<<<512,  256, 0, stream>>>(fc2w, fc2b, bufA, bufB, 4096,  0);
    fc_kernel<<<250,  256, 0, stream>>>(fc7w, fc7b, bufB, bufA, 512,   0);
    head_kernel<<<1, 64, 0, stream>>>(bufA, (float*)d_out);
}
